// Attention_20057497272967
// MI455X (gfx1250) — matrix operand order repaired and checked by reading, not compiled, not run on hardware
//
#include <hip/hip_runtime.h>
#include <hip/hip_bf16.h>

typedef __bf16 bf16;
typedef __attribute__((ext_vector_type(16))) bf16 bfrag16;
typedef __attribute__((ext_vector_type(8)))  bf16 bfrag8;
typedef __attribute__((ext_vector_type(8)))  float v8f;
typedef __attribute__((ext_vector_type(4)))  unsigned int v4u;
typedef __attribute__((ext_vector_type(8)))  int v8i;
typedef __attribute__((ext_vector_type(4)))  int v4i;

constexpr int Bb = 2, Ss = 2048, Dd = 1024, Hh = 16, HD = 64;
constexpr int Mrows = Bb * Ss;   // 4096

// ---------- WMMA helpers ----------
__device__ inline v8f wmma_bf16(bfrag16 a, bfrag16 b, v8f c) {
  // D = A(16x32 bf16) x B(32x16 bf16) + C(16x16 f32)
  return __builtin_amdgcn_wmma_f32_16x16x32_bf16(
      /*neg_a=*/false, a, /*neg_b=*/false, b,
      /*c_mod=*/(short)0, c, /*reuse_a=*/false, /*reuse_b=*/false);
}

// A-matrix 16x32 fragment: lane holds K = {klo..klo+15},
// klo = 0 (lanes 0-15) or 16 (lanes 16-31).
__device__ inline bfrag16 ld_afrag(const bf16* rowp, int klo) {
  bfrag8 lo = *(const bfrag8*)(rowp + klo);
  bfrag8 hi = *(const bfrag8*)(rowp + klo + 8);
  bfrag16 r;
#pragma unroll
  for (int e = 0; e < 8; ++e) { r[e] = lo[e]; r[e + 8] = hi[e]; }
  return r;
}

// B-matrix 32x16 fragment: per lane 16 contiguous K values (k-half in pointer).
__device__ inline bfrag16 ld_bfrag(const bf16* p) {
  bfrag8 a = *(const bfrag8*)p;
  bfrag8 b = *(const bfrag8*)(p + 8);
  bfrag16 r;
#pragma unroll
  for (int e = 0; e < 8; ++e) { r[e] = a[e]; r[e + 8] = b[e]; }
  return r;
}

// ---------- Tensor Data Mover: 2D tile (bf16) global -> LDS ----------
#if __has_builtin(__builtin_amdgcn_tensor_load_to_lds)
#define USE_TDM 1
__device__ inline void tdm_load_2d(const bf16* gptr, bf16* ldsptr,
                                   unsigned tensor_d0, unsigned tensor_d1,
                                   unsigned tile_d0, unsigned tile_d1,
                                   unsigned stride_d0) {
  unsigned long long ga = (unsigned long long)(uintptr_t)gptr;
  unsigned lds = (unsigned)(uintptr_t)ldsptr;      // low 32 bits = LDS offset
  v4u g0;
  g0.x = 1u;                                       // count=1 valid descriptor
  g0.y = lds;                                      // lds_addr
  g0.z = (unsigned)(ga & 0xFFFFFFFFu);             // global_addr[31:0]
  g0.w = (unsigned)((ga >> 32) & 0x01FFFFFFu) | (2u << 30);  // addr[56:32]|type=2
  v8i g1;
  g1[0] = (int)(1u << 16);                         // data_size=1 (2 bytes)
  g1[1] = (int)((tensor_d0 & 0xFFFFu) << 16);      // tensor_dim0[15:0]
  g1[2] = (int)(((tensor_d0 >> 16) & 0xFFFFu) |
                ((tensor_d1 & 0xFFFFu) << 16));
  g1[3] = (int)(((tensor_d1 >> 16) & 0xFFFFu) |
                ((tile_d0 & 0xFFFFu) << 16));
  g1[4] = (int)(tile_d1 & 0xFFFFu);                // tile_dim1 (tile_dim2=0)
  g1[5] = (int)stride_d0;                          // tensor_dim0_stride[31:0]
  g1[6] = 0;
  g1[7] = 0;
  v4i z4 = {0, 0, 0, 0};
#if defined(__clang_major__) && __clang_major__ >= 23
  v8i z8 = {0, 0, 0, 0, 0, 0, 0, 0};
  __builtin_amdgcn_tensor_load_to_lds(g0, g1, z4, z4, z8, 0);
#else
  __builtin_amdgcn_tensor_load_to_lds(g0, g1, z4, z4, 0);
#endif
}
#else
#define USE_TDM 0
#endif

// ---------- f32 -> bf16 conversion ----------
__global__ void cvt_f32_bf16(const float* __restrict__ src,
                             bf16* __restrict__ dst, int n) {
  int i = blockIdx.x * blockDim.x + threadIdx.x;
  int stride = gridDim.x * blockDim.x;
  for (; i < n; i += stride) dst[i] = (bf16)src[i];
}

// f32 [K,N] -> bf16 transposed [N,K]  (weights, one-time)
__global__ void cvt_f32_bf16_T(const float* __restrict__ src,
                               bf16* __restrict__ dst, int K, int N) {
  int i = blockIdx.x * blockDim.x + threadIdx.x;
  int stride = gridDim.x * blockDim.x;
  int total = K * N;
  for (; i < total; i += stride) {
    int n = i / K, k = i - n * K;
    dst[i] = (bf16)src[(size_t)k * N + n];
  }
}

// ---------- tiled bf16 WMMA GEMM:  C[M,N] = A[M,K] @ Bt[N,K]^T ----------
// TDM double-buffered: tile i+1 DMA overlaps tile i WMMA via TENSORcnt.
// MODE 0: bf16 out row-major; MODE 1: bf16 out [b,h,d,s]; MODE 2: f32 + bias.
template <int MODE>
__global__ __launch_bounds__(256) void gemm_wmma(
    const bf16* __restrict__ A, const bf16* __restrict__ Bt,
    bf16* __restrict__ outb, float* __restrict__ outf,
    const float* __restrict__ bias) {
  constexpr int K = Dd, N = Dd;
  __shared__ bf16 lA[2][64 * 32];    // 64 rows x 32 k, double buffered
  __shared__ bf16 lB[2][128 * 32];   // 128 n-rows x 32 k

  const int tid  = threadIdx.x;
  const int lane = tid & 31;
  const int wid  = tid >> 5;
  const int wm   = wid & 1;                        // 2 x 32 rows
  const int wn   = wid >> 1;                       // 4 x 32 cols
  const int m0   = blockIdx.y * 64;
  const int n0   = blockIdx.x * 128;

  v8f acc[2][2] = {};

  const int ml  = lane & 15;
  const int klo = (lane < 16) ? 0 : 16;
  const int kh  = (lane < 16) ? 0 : 16;

  constexpr int niter = K / 32;
  int buf = 0;
#if USE_TDM
  if (wid == 0) {                    // prologue: DMA tile 0 into buffer 0
    tdm_load_2d(A  + (size_t)m0 * K, lA[0], K, Mrows, 32, 64,  K);
    tdm_load_2d(Bt + (size_t)n0 * K, lB[0], K, N,     32, 128, K);
  }
#endif
  for (int it = 0; it < niter; ++it) {
#if USE_TDM
    if (wid == 0) {
      if (it + 1 < niter) {          // overlap next tile's DMA with compute
        const int kn = (it + 1) * 32;
        tdm_load_2d(A  + (size_t)m0 * K + kn, lA[buf ^ 1], K, Mrows, 32, 64,  K);
        tdm_load_2d(Bt + (size_t)n0 * K + kn, lB[buf ^ 1], K, N,     32, 128, K);
        __builtin_amdgcn_s_wait_tensorcnt(2);   // tile it's 2 ops complete
      } else {
        __builtin_amdgcn_s_wait_tensorcnt(0);
      }
    }
#else
    {  // fallback: manual staging
      const int kb = it * 32;
      int row = tid >> 2, cq = (tid & 3) * 8;      // A: 64x32
      *(bfrag8*)(lA[buf] + row * 32 + cq) =
          *(const bfrag8*)(A + (size_t)(m0 + row) * K + kb + cq);
      int rb = tid >> 1, cb = (tid & 1) * 16;      // B: 128x32
      *(bfrag8*)(lB[buf] + rb * 32 + cb) =
          *(const bfrag8*)(Bt + (size_t)(n0 + rb) * K + kb + cb);
      *(bfrag8*)(lB[buf] + rb * 32 + cb + 8) =
          *(const bfrag8*)(Bt + (size_t)(n0 + rb) * K + kb + cb + 8);
    }
#endif
    __syncthreads();

    bfrag16 af[2], bfg[2];
#pragma unroll
    for (int i = 0; i < 2; ++i)
      af[i] = ld_afrag(lA[buf] + (wm * 32 + i * 16 + ml) * 32, klo);
#pragma unroll
    for (int j = 0; j < 2; ++j)
      bfg[j] = ld_bfrag(lB[buf] + (wn * 32 + j * 16 + ml) * 32 + kh);
#pragma unroll
    for (int i = 0; i < 2; ++i)
#pragma unroll
      for (int j = 0; j < 2; ++j)
        acc[i][j] = wmma_bf16(af[i], bfg[j], acc[i][j]);
    __syncthreads();                 // all reads of buf done before its reuse
    buf ^= 1;
  }

  // C layout: VGPR r -> row (r / r+8 by lane half), lane%16 -> col
#pragma unroll
  for (int i = 0; i < 2; ++i) {
#pragma unroll
    for (int j = 0; j < 2; ++j) {
      const int col = n0 + wn * 32 + j * 16 + ml;
      const float bval = (MODE == 2) ? bias[col] : 0.f;
#pragma unroll
      for (int r = 0; r < 8; ++r) {
        const int rowm = (lane < 16) ? r : r + 8;
        const int row  = m0 + wm * 32 + i * 16 + rowm;
        const float v  = acc[i][j][r];
        if (MODE == 0) {
          outb[(size_t)row * N + col] = (bf16)v;
        } else if (MODE == 1) {
          const int bb = row >> 11, s = row & (Ss - 1);
          const int h = col >> 6, d = col & (HD - 1);
          outb[(((size_t)bb * Hh + h) * HD + d) * Ss + s] = (bf16)v;
        } else {
          outf[(size_t)row * N + col] = v + bval;
        }
      }
    }
  }
}

// ---------- flash attention, one wave per 32-row query tile ----------
// K/V fragments are loaded once per kv-block and reused by both 16-row
// query sub-tiles (2x arithmetic intensity vs 16-row tiles).
__global__ __launch_bounds__(128) void attn_wmma(
    const bf16* __restrict__ Q, const bf16* __restrict__ Km,
    const bf16* __restrict__ Vt, bf16* __restrict__ ctx) {
  __shared__ bf16 Plds[4][32 * 32];                // per-wave P staging

  const int lane = threadIdx.x & 31;
  const int wid  = threadIdx.x >> 5;
  const int task = blockIdx.x * 4 + wid;           // 2048 tasks
  const int qt = task & 63;                        // 64 q-tiles of 32 rows
  const int bh = task >> 6;
  const int h = bh & (Hh - 1), b = bh >> 4;

  const int ml  = lane & 15;
  const int klo = (lane < 16) ? 0 : 16;
  const int khv = (lane < 16) ? 0 : 16;

  // Q fragments: 2 sub-tiles x (two 16x32 frags covering d=0..63)
  bfrag16 aq[2][2];
#pragma unroll
  for (int u = 0; u < 2; ++u) {
    const int qrow = qt * 32 + u * 16 + ml;
    const bf16* qp = Q + ((size_t)(b * Ss + qrow)) * Dd + h * HD;
    aq[u][0] = ld_afrag(qp, klo);
    aq[u][1] = ld_afrag(qp + 32, klo);
  }

  v8f O[2][4] = {};                                // 32x64 f32 output
  float mrow[2][8], lrow[2][8];
#pragma unroll
  for (int u = 0; u < 2; ++u)
#pragma unroll
    for (int r = 0; r < 8; ++r) { mrow[u][r] = -__builtin_inff(); lrow[u][r] = 0.f; }

  bf16* P = &Plds[wid][0];
  const int nblk = qt + 1;                         // causal: kv blocks of 32

  for (int j = 0; j < nblk; ++j) {
    const int kv0 = j * 32;

    // ---- K fragments (shared by both q sub-tiles) ----
    const bf16* kp0 = Km + ((size_t)(b * Ss + kv0 + ml)) * Dd + h * HD + khv;
    const bf16* kp1 = Km + ((size_t)(b * Ss + kv0 + 16 + ml)) * Dd + h * HD + khv;
    bfrag16 bk00 = ld_bfrag(kp0),      bk01 = ld_bfrag(kp0 + 32);
    bfrag16 bk10 = ld_bfrag(kp1),      bk11 = ld_bfrag(kp1 + 32);

    // ---- scores: 2 sub-tiles x 2 key-tiles ----
    v8f sc[2][2];
#pragma unroll
    for (int u = 0; u < 2; ++u) {
      v8f s = {};
      s = wmma_bf16(aq[u][0], bk00, s);
      sc[u][0] = wmma_bf16(aq[u][1], bk01, s);
      v8f t = {};
      t = wmma_bf16(aq[u][0], bk10, t);
      sc[u][1] = wmma_bf16(aq[u][1], bk11, t);
    }

    // ---- causal mask + online softmax per sub-tile ----
#pragma unroll
    for (int u = 0; u < 2; ++u) {
      float p0a[8], p1a[8], corr[8];
#pragma unroll
      for (int r = 0; r < 8; ++r) {
        const int rowm = (lane < 16) ? r : r + 8;
        const int qg = qt * 32 + u * 16 + rowm;
        float v0 = sc[u][0][r] * 0.125f;           // 1/sqrt(64)
        if (kv0 + ml > qg) v0 = -__builtin_inff();
        float v1 = sc[u][1][r] * 0.125f;
        if (kv0 + 16 + ml > qg) v1 = -__builtin_inff();
        float mx = fmaxf(v0, v1);
#pragma unroll
        for (int o = 8; o > 0; o >>= 1) mx = fmaxf(mx, __shfl_xor(mx, o, 16));
        const float mnew = fmaxf(mrow[u][r], mx);
        const float p0 = __expf(v0 - mnew);
        const float p1 = __expf(v1 - mnew);
        float ps = p0 + p1;
#pragma unroll
        for (int o = 8; o > 0; o >>= 1) ps += __shfl_xor(ps, o, 16);
        const float c = __expf(mrow[u][r] - mnew);
        lrow[u][r] = lrow[u][r] * c + ps;
        mrow[u][r] = mnew;
        p0a[r] = p0; p1a[r] = p1; corr[r] = c;
      }
#pragma unroll
      for (int t = 0; t < 4; ++t)
#pragma unroll
        for (int r = 0; r < 8; ++r) O[u][t][r] *= corr[r];

      // transpose P (C layout -> A layout) through per-wave LDS
#pragma unroll
      for (int r = 0; r < 8; ++r) {
        const int rowm = (lane < 16) ? r : r + 8;
        P[(u * 16 + rowm) * 32 + ml]      = (bf16)p0a[r];
        P[(u * 16 + rowm) * 32 + 16 + ml] = (bf16)p1a[r];
      }
    }
    asm volatile("s_wait_dscnt 0" ::: "memory");   // wave-local LDS RAW fence
    const bfrag16 ap0 = ld_afrag(P + ml * 32, klo);
    const bfrag16 ap1 = ld_afrag(P + (16 + ml) * 32, klo);

    // ---- O += P @ V : V fragments shared by both sub-tiles ----
#pragma unroll
    for (int t = 0; t < 4; ++t) {
      const bf16* vp =
          Vt + (((size_t)b * Hh + h) * HD + t * 16 + ml) * Ss + kv0 + khv;
      const bfrag16 bv = ld_bfrag(vp);
      O[0][t] = wmma_bf16(ap0, bv, O[0][t]);
      O[1][t] = wmma_bf16(ap1, bv, O[1][t]);
    }
    asm volatile("s_wait_dscnt 0" ::: "memory");   // reads done before next P write
  }

  // ---- normalize and store context (bf16, [b,s,h*64+d]) ----
#pragma unroll
  for (int u = 0; u < 2; ++u)
#pragma unroll
    for (int r = 0; r < 8; ++r) lrow[u][r] = 1.f / lrow[u][r];
#pragma unroll
  for (int u = 0; u < 2; ++u) {
#pragma unroll
    for (int t = 0; t < 4; ++t) {
#pragma unroll
      for (int r = 0; r < 8; ++r) {
        const int rowm = (lane < 16) ? r : r + 8;
        const int d = t * 16 + ml;
        ctx[((size_t)(b * Ss + qt * 32 + u * 16 + rowm)) * Dd + h * HD + d] =
            (bf16)(O[u][t][r] * lrow[u][r]);
      }
    }
  }
}

// ---------- host entry ----------
extern "C" void kernel_launch(void* const* d_in, const int* in_sizes, int n_in,
                              void* d_out, int out_size, void* d_ws,
                              size_t ws_size, hipStream_t stream) {
  const float* x  = (const float*)d_in[0];
  const float* wq = (const float*)d_in[1];
  const float* wk = (const float*)d_in[2];
  const float* wv = (const float*)d_in[3];
  const float* wo = (const float*)d_in[4];
  const float* bo = (const float*)d_in[5];

  bf16* ws = (bf16*)d_ws;
  const size_t szX = (size_t)Mrows * Dd;   // 4096*1024
  const size_t szW = (size_t)Dd * Dd;      // 1024*1024
  bf16* xbf = ws;                          // [M,K]
  bf16* wqt = xbf + szX;                   // weights, transposed [N,K] bf16
  bf16* wkt = wqt + szW;
  bf16* wvt = wkt + szW;
  bf16* wot = wvt + szW;
  bf16* Qb  = wot + szW;                   // [b,s,h*64+d]
  bf16* Kb  = Qb + szX;
  bf16* Vtb = Kb + szX;                    // [b,h,d,s]
  bf16* Cb  = Vtb + szX;                   // attention context

  cvt_f32_bf16  <<<512, 256, 0, stream>>>(x,  xbf, (int)szX);
  cvt_f32_bf16_T<<<256, 256, 0, stream>>>(wq, wqt, Dd, Dd);
  cvt_f32_bf16_T<<<256, 256, 0, stream>>>(wk, wkt, Dd, Dd);
  cvt_f32_bf16_T<<<256, 256, 0, stream>>>(wv, wvt, Dd, Dd);
  cvt_f32_bf16_T<<<256, 256, 0, stream>>>(wo, wot, Dd, Dd);

  dim3 g(Dd / 128, Mrows / 64);            // (8, 64)
  gemm_wmma<0><<<g, 256, 0, stream>>>(xbf, wqt, Qb,  nullptr, nullptr);
  gemm_wmma<0><<<g, 256, 0, stream>>>(xbf, wkt, Kb,  nullptr, nullptr);
  gemm_wmma<1><<<g, 256, 0, stream>>>(xbf, wvt, Vtb, nullptr, nullptr);

  attn_wmma<<<(Bb * Hh * (Ss / 32)) / 4, 128, 0, stream>>>(Qb, Kb, Vtb, Cb);

  gemm_wmma<2><<<g, 256, 0, stream>>>(Cb, wot, nullptr, (float*)d_out, bo);
}
